// TtViTSelfAttention_39367670235206
// MI455X (gfx1250) — matrix operand order repaired and checked by reading, not compiled, not run on hardware
//
#include <hip/hip_runtime.h>
#include <hip/hip_bf16.h>

// ---------------------------------------------------------------------------
// Types for CDNA5 WMMA (wave32): v_wmma_f32_16x16x32_bf16
// ---------------------------------------------------------------------------
typedef __attribute__((ext_vector_type(16))) __bf16 bf16x16;
typedef __attribute__((ext_vector_type(8)))  __bf16 bf16x8;
typedef __attribute__((ext_vector_type(8)))  float  f32x8;

#define NUM_HEADS 16
#define HEAD_DIM  64
#define DMODEL    1024
#define BATCH     32
#define SEQ       577
#define SP        608    // SEQ padded to multiple of 32 (19 chunks of 32)
#define QTILES    37     // ceil(577/16) query tiles actually written
#define LDSW_STRIDE 1032 // 1024 + 8 pad (ushorts): de-conflicts LDS banks

// ---------------------------------------------------------------------------
// Helpers
// ---------------------------------------------------------------------------
__device__ __forceinline__ unsigned short f2bf(float f) {
  unsigned int u = __float_as_uint(f);
  u += 0x7FFFu + ((u >> 16) & 1u);          // round to nearest even
  return (unsigned short)(u >> 16);
}

// A-matrix 16x32 bf16 fragment: lane holds row m = lane&15.
// element s -> K = hi*16 + s (contiguous, matching the B fragments)
// => two contiguous 16B chunks per lane at k = hi*16 and k = hi*16 + 8.
__device__ __forceinline__ bf16x16 load_a_rowptr(const unsigned short* prow) {
  int hi8 = ((threadIdx.x >> 4) & 1) << 4;
  bf16x8 lo = *(const bf16x8*)(prow + hi8);
  bf16x8 hh = *(const bf16x8*)(prow + 8 + hi8);
  bf16x16 r;
#pragma unroll
  for (int i = 0; i < 8; ++i) { r[i] = lo[i]; r[i + 8] = hh[i]; }
  return r;
}

// B-matrix 32x16 bf16 fragment: lane holds column n = lane&15;
// lanes 0-15 hold K=0..15 (pairs 2v,2v+1), lanes 16-31 hold K=16..31.
// Memory layout: element (k, n) at base + n*stride + k  => one 32B load/lane.
__device__ __forceinline__ bf16x16 load_b_frag(const unsigned short* base, int stride) {
  int lane = threadIdx.x & 31;
  const unsigned short* p = base + (size_t)(lane & 15) * stride + ((lane >> 4) << 4);
  return *(const bf16x16*)p;
}

__device__ __forceinline__ f32x8 wmma_bf16(bf16x16 a, bf16x16 b, f32x8 c) {
  return __builtin_amdgcn_wmma_f32_16x16x32_bf16(false, a, false, b, (short)0, c,
                                                 false, false);
}

// ---------------------------------------------------------------------------
// Kernel 1: fp32 -> bf16 conversion (vectorized x4, grid-stride)
// ---------------------------------------------------------------------------
__global__ void __launch_bounds__(256) cvt_f32_bf16(const float* __restrict__ s,
                                                    unsigned short* __restrict__ d,
                                                    size_t n4) {
  size_t i = (size_t)blockIdx.x * blockDim.x + threadIdx.x;
  size_t stride = (size_t)gridDim.x * blockDim.x;
  for (; i < n4; i += stride) {
    float4 f = ((const float4*)s)[i];
    ushort4 o;
    o.x = f2bf(f.x); o.y = f2bf(f.y); o.z = f2bf(f.z); o.w = f2bf(f.w);
    ((ushort4*)d)[i] = o;
  }
}

// ---------------------------------------------------------------------------
// One GEMM K-step: B-fragments just-in-time from LDS (one live at a time ->
// low register pressure; this is the variant that scheduled cleanly), 8 WMMAs.
// ---------------------------------------------------------------------------
__device__ __forceinline__ void g_step(const bf16x16& a0, const bf16x16& a1,
                                       const unsigned short* ldsW, int j0,
                                       int nl, int hi, f32x8* acc) {
#pragma unroll
  for (int t = 0; t < 4; ++t) {
    bf16x16 bfr =
        *(const bf16x16*)&ldsW[(size_t)(t * 16 + nl) * LDSW_STRIDE + j0 + (hi << 4)];
    acc[t]     = wmma_bf16(a0, bfr, acc[t]);
    acc[4 + t] = wmma_bf16(a1, bfr, acc[4 + t]);
  }
}

// ---------------------------------------------------------------------------
// Kernel 2: fused QKV projection  Y = X @ W^T + b  (bf16 WMMA, f32 accum)
// grid = (76, 16, 3), block = 256 (8 waves), dynamic LDS = 132,096 B.
// Block stages one whole 64-column W strip (one head) into LDS once using
// GLOBAL_LOAD_ASYNC_TO_LDS_B128 (ASYNCcnt-tracked DMA path, no VGPR
// round-trip); the 8 waves each own a different 32-row group of the strip.
// A-fragments ping-pong from global; B-fragments JIT from LDS.
// Q,K -> [B][H][SP][64] bf16 ; V -> transposed [B][H][64][SP] bf16.
// Rows s >= 577 are written as zero (pads for the attention kernel).
// ---------------------------------------------------------------------------
__global__ void __launch_bounds__(256) qkv_gemm(
    const unsigned short* __restrict__ Xbf,
    const unsigned short* __restrict__ Wqb, const unsigned short* __restrict__ Wkb,
    const unsigned short* __restrict__ Wvb,
    const float* __restrict__ bq, const float* __restrict__ bk,
    const float* __restrict__ bv,
    unsigned short* __restrict__ Qb, unsigned short* __restrict__ Kb,
    unsigned short* __restrict__ Vtb) {
  extern __shared__ unsigned short ldsW[];   // [64][LDSW_STRIDE] ushorts

  const int z = blockIdx.z;
  const unsigned short* W = (z == 0) ? Wqb : (z == 1) ? Wkb : Wvb;
  const float* bias = (z == 0) ? bq : (z == 1) ? bk : bv;

  const int tid  = threadIdx.x;
  const int wave = tid >> 5;
  const int lane = tid & 31;
  const int nl = lane & 15, hi = lane >> 4;

  const int i0 = blockIdx.y * 64;            // 64-wide => exactly one head

  // ---- async stage: W rows i0..i0+63 (128KB) -> LDS, once ----
  // GLOBAL_LOAD_ASYNC_TO_LDS_B128 (GV mode): vdst = per-lane LDS byte offset
  // (low 32 bits of the LDS pointer, per ISA addr[31:0] truncation),
  // vaddr = 64-bit global address, "off" = no SADDR. Tracked by ASYNCcnt.
  const unsigned short* wsrc = W + (size_t)i0 * DMODEL;
#pragma unroll
  for (int c = 0; c < 8192; c += 256) {      // 8192 chunks of 16B
    const int cc = c + tid;
    const int n = cc >> 7, off = (cc & 127) << 3;   // ushort offsets
    unsigned ldsoff = (unsigned)(size_t)&ldsW[(size_t)n * LDSW_STRIDE + off];
    const unsigned short* gp = &wsrc[(size_t)n * DMODEL + off];
    asm volatile("global_load_async_to_lds_b128 %0, %1, off"
                 :: "v"(ldsoff), "v"(gp)
                 : "memory");
  }
  asm volatile("s_wait_asynccnt 0x0" ::: "memory");
  __syncthreads();

  // ---- per-wave 32x64 tile ----
  const int g  = blockIdx.x * 256 + wave * 32;  // global padded row (19456 total)
  const int b  = g / SP;
  const int s0 = g % SP;                         // 608 % 32 == 0: no straddle

  int sr0 = s0 + nl;       if (sr0 > SEQ - 1) sr0 = SEQ - 1;
  int sr1 = s0 + 16 + nl;  if (sr1 > SEQ - 1) sr1 = SEQ - 1;
  const unsigned short* xrow0 = Xbf + ((size_t)b * SEQ + sr0) * DMODEL;
  const unsigned short* xrow1 = Xbf + ((size_t)b * SEQ + sr1) * DMODEL;

  f32x8 acc[8] = {};

  // A-fragment ping-pong over 32 K-steps (j0 = 0,32,...,992)
  bf16x16 aA0 = load_a_rowptr(xrow0);
  bf16x16 aA1 = load_a_rowptr(xrow1);
  bf16x16 aB0, aB1;
  for (int j0 = 0; j0 < DMODEL - 64; j0 += 64) {
    aB0 = load_a_rowptr(xrow0 + j0 + 32);
    aB1 = load_a_rowptr(xrow1 + j0 + 32);
    g_step(aA0, aA1, ldsW, j0, nl, hi, acc);
    aA0 = load_a_rowptr(xrow0 + j0 + 64);
    aA1 = load_a_rowptr(xrow1 + j0 + 64);
    g_step(aB0, aB1, ldsW, j0 + 32, nl, hi, acc);
  }
  aB0 = load_a_rowptr(xrow0 + DMODEL - 32);
  aB1 = load_a_rowptr(xrow1 + DMODEL - 32);
  g_step(aA0, aA1, ldsW, DMODEL - 64, nl, hi, acc);
  g_step(aB0, aB1, ldsW, DMODEL - 32, nl, hi, acc);

  const int h = i0 >> 6;
  const size_t headRM = ((size_t)b * NUM_HEADS + h) * SP * HEAD_DIM;  // Q,K
  const size_t headCM = ((size_t)b * NUM_HEADS + h) * HEAD_DIM * SP;  // V^T
#pragma unroll
  for (int gg = 0; gg < 2; ++gg) {
#pragma unroll
    for (int t = 0; t < 4; ++t) {
      const int col = t * 16 + nl;               // dd within head
      const float bb = bias[i0 + col];
#pragma unroll
      for (int r = 0; r < 8; ++r) {
        const int s = s0 + gg * 16 + r + 8 * hi; // C layout: row = r + 8*hi
        const unsigned short v =
            (s < SEQ) ? f2bf(acc[gg * 4 + t][r] + bb) : (unsigned short)0;
        if (z == 0)      Qb [headRM + (size_t)s * HEAD_DIM + col] = v;
        else if (z == 1) Kb [headRM + (size_t)s * HEAD_DIM + col] = v;
        else             Vtb[headCM + (size_t)col * SP + s]       = v;
      }
    }
  }
}

// ---------------------------------------------------------------------------
// Attention: K-fragments for one 32-key chunk (V is loaded JIT in process)
// ---------------------------------------------------------------------------
struct KF { bf16x16 k00, k01, k10, k11; };

__device__ __forceinline__ void k_load(KF& f, const unsigned short* Kh, int kb) {
  f.k00 = load_b_frag(Kh + (size_t)kb * HEAD_DIM, HEAD_DIM);
  f.k01 = load_b_frag(Kh + (size_t)kb * HEAD_DIM + 32, HEAD_DIM);
  f.k10 = load_b_frag(Kh + (size_t)(kb + 16) * HEAD_DIM, HEAD_DIM);
  f.k11 = load_b_frag(Kh + (size_t)(kb + 16) * HEAD_DIM + 32, HEAD_DIM);
}

// One 32-key chunk of flash attention for a 16-query tile.
__device__ __forceinline__ void kv_process(const KF& f, const unsigned short* Vh,
                                           int kb,
                                           const bf16x16& qa0, const bf16x16& qa1,
                                           f32x8* ctx, float* row_m, float* row_l,
                                           unsigned short* pp, int nl, int hi) {
  // ---- V-fragment loads first: overlap the score WMMAs + softmax ----
  bf16x16 vf0 = load_b_frag(Vh + (size_t)( 0) * SP + kb, SP);
  bf16x16 vf1 = load_b_frag(Vh + (size_t)(16) * SP + kb, SP);
  bf16x16 vf2 = load_b_frag(Vh + (size_t)(32) * SP + kb, SP);
  bf16x16 vf3 = load_b_frag(Vh + (size_t)(48) * SP + kb, SP);

  // ---- scores: two 16x16 tiles, K = 64 over head dim ----
  f32x8 sc0 = {}, sc1 = {};
  sc0 = wmma_bf16(qa0, f.k00, sc0);
  sc0 = wmma_bf16(qa1, f.k01, sc0);
  sc1 = wmma_bf16(qa0, f.k10, sc1);
  sc1 = wmma_bf16(qa1, f.k11, sc1);

  // ---- scale + key mask (columns live per-lane) ----
  const bool ok0 = (kb + nl) < SEQ;
  const bool ok1 = (kb + 16 + nl) < SEQ;
#pragma unroll
  for (int r = 0; r < 8; ++r) {
    sc0[r] = ok0 ? sc0[r] * 0.125f : -3.0e38f;
    sc1[r] = ok1 ? sc1[r] * 0.125f : -3.0e38f;
  }

  // ---- online softmax: rows are spread over 16-lane halves ----
  float alpha[8];
#pragma unroll
  for (int r = 0; r < 8; ++r) {
    float tm = fmaxf(sc0[r], sc1[r]);
#pragma unroll
    for (int m = 8; m >= 1; m >>= 1) tm = fmaxf(tm, __shfl_xor(tm, m, 32));
    const float nm = fmaxf(row_m[r], tm);
    alpha[r] = __expf(row_m[r] - nm);
    row_m[r] = nm;
    const float p0 = __expf(sc0[r] - nm);
    const float p1 = __expf(sc1[r] - nm);
    sc0[r] = p0; sc1[r] = p1;
    float rs = p0 + p1;
#pragma unroll
    for (int m = 8; m >= 1; m >>= 1) rs += __shfl_xor(rs, m, 32);
    row_l[r] = row_l[r] * alpha[r] + rs;
  }

  // ---- rescale running context ----
#pragma unroll
  for (int t = 0; t < 4; ++t)
#pragma unroll
    for (int r = 0; r < 8; ++r) ctx[t][r] *= alpha[r];

  // ---- C-layout P -> LDS (row-major 16x32 bf16) -> A-fragment ----
#pragma unroll
  for (int r = 0; r < 8; ++r) {
    const int mrow = r + 8 * hi;
    pp[mrow * 32 + nl]      = f2bf(sc0[r]);
    pp[mrow * 32 + 16 + nl] = f2bf(sc1[r]);
  }
  const bf16x16 pa = load_a_rowptr(pp + nl * 32);   // per-wave, in-order LDS

  // ---- ctx += P @ V ----
  ctx[0] = wmma_bf16(pa, vf0, ctx[0]);
  ctx[1] = wmma_bf16(pa, vf1, ctx[1]);
  ctx[2] = wmma_bf16(pa, vf2, ctx[2]);
  ctx[3] = wmma_bf16(pa, vf3, ctx[3]);
}

// ---------------------------------------------------------------------------
// Kernel 3: flash-style attention. grid = (ceil(37/8), 16, 32), block = 256.
// Each wave owns one 16-query tile of one (b,h). The 19-chunk key loop
// ping-pongs only the K-fragments (low pressure); V loads JIT inside process.
// ---------------------------------------------------------------------------
__global__ void __launch_bounds__(256) attn_fwd(
    const unsigned short* __restrict__ Qb, const unsigned short* __restrict__ Kb,
    const unsigned short* __restrict__ Vtb, float* __restrict__ out) {
  __shared__ __align__(64) unsigned short pstage[8][16 * 32];

  const int wave = threadIdx.x >> 5;
  const int lane = threadIdx.x & 31;
  const int nl = lane & 15, hi = lane >> 4;

  const int qt = blockIdx.x * 8 + wave;
  if (qt >= QTILES) return;                      // wave-uniform: EXEC stays full
  const int h = blockIdx.y, b = blockIdx.z;
  const int s0 = qt * 16;

  const unsigned short* Qh = Qb  + ((size_t)b * NUM_HEADS + h) * SP * HEAD_DIM;
  const unsigned short* Kh = Kb  + ((size_t)b * NUM_HEADS + h) * SP * HEAD_DIM;
  const unsigned short* Vh = Vtb + ((size_t)b * NUM_HEADS + h) * HEAD_DIM * SP;

  // Q A-fragments (d = 0..31, 32..63), held in registers for the whole loop
  const unsigned short* qrow = Qh + (size_t)(s0 + nl) * HEAD_DIM;
  const bf16x16 qa0 = load_a_rowptr(qrow);
  const bf16x16 qa1 = load_a_rowptr(qrow + 32);

  f32x8 ctx[4] = {};
  float row_m[8], row_l[8];
#pragma unroll
  for (int r = 0; r < 8; ++r) { row_m[r] = -3.0e38f; row_l[r] = 0.0f; }

  unsigned short* pp = &pstage[wave][0];

  // 19 chunks of 32 keys: prologue + 9 ping-pong pairs + peeled tail
  KF kA, kB;
  k_load(kA, Kh, 0);
  int kb = 0;
  for (int it = 0; it < 9; ++it) {
    k_load(kB, Kh, kb + 32);
    kv_process(kA, Vh, kb, qa0, qa1, ctx, row_m, row_l, pp, nl, hi);
    k_load(kA, Kh, kb + 64);
    kv_process(kB, Vh, kb + 32, qa0, qa1, ctx, row_m, row_l, pp, nl, hi);
    kb += 64;
  }
  kv_process(kA, Vh, SP - 32, qa0, qa1, ctx, row_m, row_l, pp, nl, hi);

  // ---- epilogue: divide by softmax denom, write [B,S,1024] fp32 ----
#pragma unroll
  for (int t = 0; t < 4; ++t) {
#pragma unroll
    for (int r = 0; r < 8; ++r) {
      const int s = s0 + r + 8 * hi;
      if (s < SEQ) {
        out[((size_t)b * SEQ + s) * DMODEL + (h << 6) + t * 16 + nl] =
            ctx[t][r] / row_l[r];
      }
    }
  }
}

// ---------------------------------------------------------------------------
// Host launcher
// ---------------------------------------------------------------------------
extern "C" void kernel_launch(void* const* d_in, const int* in_sizes, int n_in,
                              void* d_out, int out_size, void* d_ws, size_t ws_size,
                              hipStream_t stream) {
  const float* X  = (const float*)d_in[0];
  const float* Wq = (const float*)d_in[1];
  const float* bq = (const float*)d_in[2];
  const float* Wk = (const float*)d_in[3];
  const float* bk = (const float*)d_in[4];
  const float* Wv = (const float*)d_in[5];
  const float* bv = (const float*)d_in[6];
  float* out = (float*)d_out;

  // Workspace layout (bytes, 256-aligned):
  //   Xbf   : 18464*1024*2           = 37,814,272
  //   Wq/k/v: 3 * 1024*1024*2        =  6,291,456
  //   Qb,Kb : [32][16][608][64] bf16 = 39,845,888 each
  //   Vtb   : [32][16][64][608] bf16 = 39,845,888
  char* ws = (char*)d_ws;
  unsigned short* Xbf = (unsigned short*)(ws);
  unsigned short* Wqb = (unsigned short*)(ws + 37814272);
  unsigned short* Wkb = (unsigned short*)(ws + 37814272 + 2097152);
  unsigned short* Wvb = (unsigned short*)(ws + 37814272 + 2 * 2097152);
  unsigned short* Qb  = (unsigned short*)(ws + 44105728);
  unsigned short* Kb  = (unsigned short*)(ws + 44105728 + 39845888);
  unsigned short* Vtb = (unsigned short*)(ws + 44105728 + 2 * 39845888);

  // 1) fp32 -> bf16
  const size_t nX = (size_t)BATCH * SEQ * DMODEL;   // 18,907,136 (mult of 4)
  const size_t nW = (size_t)DMODEL * DMODEL;        //  1,048,576
  {
    int gx = (int)((nX / 4 + 255) / 256);
    cvt_f32_bf16<<<gx, 256, 0, stream>>>(X, Xbf, nX / 4);
    int gw = (int)((nW / 4 + 255) / 256);
    cvt_f32_bf16<<<gw, 256, 0, stream>>>(Wq, Wqb, nW / 4);
    cvt_f32_bf16<<<gw, 256, 0, stream>>>(Wk, Wkb, nW / 4);
    cvt_f32_bf16<<<gw, 256, 0, stream>>>(Wv, Wvb, nW / 4);
  }

  // 2) QKV projection: W strip async-staged into LDS (132,096 B dynamic)
  qkv_gemm<<<dim3(76, 16, 3), 256, 64 * LDSW_STRIDE * 2, stream>>>(
      Xbf, Wqb, Wkb, Wvb, bq, bk, bv, Qb, Kb, Vtb);

  // 3) attention, K-fragment ping-pong + JIT V
  attn_fwd<<<dim3((QTILES + 7) / 8, NUM_HEADS, BATCH), 256, 0, stream>>>(
      Qb, Kb, Vtb, out);
}
